// RandomProjectionQuantizer_31241592111229
// MI455X (gfx1250) — matrix operand order repaired and checked by reading, not compiled, not run on hardware
//
#include <hip/hip_runtime.h>

// ---------------------------------------------------------------------------
// RandomProjectionQuantizer for MI455X (gfx1250, wave32, WMMA + async LDS).
//   targets = X @ W^T                       (8192x320)x(320x512)
//   idx     = argmin_j ( c2[j] - 2 * targets . cb[j] )   over 8192 codes
// fp32 inputs split into bf16 hi/lo pairs; 3x v_wmma_f32_16x16x32_bf16 per
// K-chunk (hi*hi + hi*lo + lo*hi) gives ~fp32 accuracy at bf16 rates.
// Distance matrix (268MB) never materialized: argmin fused in the epilogue.
// Workgroup = 4 waves x 16 tokens = 64 tokens; all waves scan the SAME code
// stream so codebook B-fragments hit the WGP cache (4x less L2 traffic).
// Target panels are staged to LDS with global_load_async_to_lds_b128.
// ---------------------------------------------------------------------------

typedef __attribute__((ext_vector_type(16))) __bf16 v16bf;
typedef __attribute__((ext_vector_type(8)))  float  v8f;
typedef __bf16 bf16_t;

#define TOKENS 8192
#define DIN    320
#define CD     512
#define NC     8192

static __device__ __forceinline__ bf16_t f2bf(float f) {
  unsigned u = __builtin_bit_cast(unsigned, f);
  unsigned r = u + 0x7FFFu + ((u >> 16) & 1u);            // round-to-nearest-even
  unsigned short h = (unsigned short)(r >> 16);
  return __builtin_bit_cast(bf16_t, h);
}
static __device__ __forceinline__ float bf2f(bf16_t b) {
  unsigned short h = __builtin_bit_cast(unsigned short, b);
  return __builtin_bit_cast(float, ((unsigned)h) << 16);
}

static __device__ __forceinline__ v8f wmma_bf16(v16bf a, v16bf b, v8f c) {
  // 8-arg form: (neg_a, A, neg_b, B, c_mod, C, reuse_a, reuse_b)
  return __builtin_amdgcn_wmma_f32_16x16x32_bf16(false, a, false, b, (short)0, c,
                                                 false, false);
}

// A-operand fragment: 16x32 bf16 tile of row-major [rows x ld] at (row0, k0).
// lanes 0-15: row m, K = k0..k0+15 (contiguous); lanes 16-31: K += 16.
static __device__ __forceinline__ v16bf load_frag_a(const bf16_t* base, int ld,
                                                    int row0, int k0, int lane) {
  const int m  = lane & 15;
  const int kk = k0 + ((lane >> 4) << 4);
  const bf16_t* p = base + (size_t)(row0 + m) * ld + kk;
  v16bf a;
  *(uint4*)&a       = *(const uint4*)(p);
  *((uint4*)&a + 1) = *(const uint4*)(p + 8);
  return a;
}

// B-operand fragment: B(k,n) = M[col0+n][k] for row-major [cols x ld].
// lane n<16: column n, K = k0..k0+15 (contiguous); lanes 16-31: K += 16.
static __device__ __forceinline__ v16bf load_frag_b(const bf16_t* base, int ld,
                                                    int col0, int k0, int lane) {
  const int n  = lane & 15;
  const int kk = k0 + ((lane >> 4) << 4);
  const bf16_t* p = base + (size_t)(col0 + n) * ld + kk;
  v16bf b;
  *(uint4*)&b       = *(const uint4*)(p);
  *((uint4*)&b + 1) = *(const uint4*)(p + 8);
  return b;
}

// --------------------------- fp32 -> bf16 hi/lo split -----------------------
__global__ void rpq_split(const float* __restrict__ x, bf16_t* __restrict__ hi,
                          bf16_t* __restrict__ lo, int n) {
  int i = blockIdx.x * blockDim.x + threadIdx.x;
  if (i < n) {
    float v = x[i];
    bf16_t h = f2bf(v);
    hi[i] = h;
    lo[i] = f2bf(v - bf2f(h));
  }
}

// ------------------------------- ||c||^2 ------------------------------------
__global__ void rpq_c2(const float* __restrict__ cb, float* __restrict__ c2) {
  int code = blockIdx.x * blockDim.x + threadIdx.x;
  if (code < NC) {
    const float* p = cb + (size_t)code * CD;
    float s = 0.f;
    for (int k = 0; k < CD; ++k) { float v = p[k]; s += v * v; }
    c2[code] = s;
  }
}

// ------------------- projection GEMM (one wave per 16x16 tile) --------------
__global__ void __launch_bounds__(32) rpq_proj(
    const bf16_t* __restrict__ Xhi, const bf16_t* __restrict__ Xlo,
    const bf16_t* __restrict__ Whi, const bf16_t* __restrict__ Wlo,
    bf16_t* __restrict__ Thi, bf16_t* __restrict__ Tlo) {
  const int db   = blockIdx.x * 16;   // output-dim tile
  const int tb   = blockIdx.y * 16;   // token tile
  const int lane = threadIdx.x & 31;

  v8f acc = {0.f, 0.f, 0.f, 0.f, 0.f, 0.f, 0.f, 0.f};
#pragma unroll
  for (int kc = 0; kc < DIN / 32; ++kc) {
    const int k0 = kc * 32;
    v16bf ah = load_frag_a(Xhi, DIN, tb, k0, lane);
    v16bf al = load_frag_a(Xlo, DIN, tb, k0, lane);
    v16bf bh = load_frag_b(Whi, DIN, db, k0, lane);
    v16bf bl = load_frag_b(Wlo, DIN, db, k0, lane);
    acc = wmma_bf16(ah, bh, acc);
    acc = wmma_bf16(ah, bl, acc);
    acc = wmma_bf16(al, bh, acc);
  }
  // C/D layout: element r of acc -> row = r + 8*(lane>=16), col = lane&15
  const int n  = lane & 15;
  const int mo = (lane >> 4) << 3;
#pragma unroll
  for (int r = 0; r < 8; ++r) {
    const size_t o = (size_t)(tb + mo + r) * CD + (db + n);
    float v  = acc[r];
    bf16_t h = f2bf(v);
    Thi[o] = h;
    Tlo[o] = f2bf(v - bf2f(h));
  }
}

// ---- distance GEMM + fused argmin: 4 waves x 16 tokens, shared code scan ---
__global__ void __launch_bounds__(128) rpq_dist_argmin(
    const bf16_t* __restrict__ Thi, const bf16_t* __restrict__ Tlo,
    const bf16_t* __restrict__ CBhi, const bf16_t* __restrict__ CBlo,
    const float* __restrict__ c2, float* __restrict__ out) {
  __shared__ __align__(16) bf16_t sAhi[4][16 * CD];   // 64 KB
  __shared__ __align__(16) bf16_t sAlo[4][16 * CD];   // 64 KB
  __shared__ float sVal[4][16 * 16];
  __shared__ int   sIdx[4][16 * 16];

  const int tb   = blockIdx.x * 64;
  const int tid  = threadIdx.x;
  const int wave = tid >> 5;
  const int lane = tid & 31;

  // --- stage each wave's 16xCD target panel (hi+lo) via async loads to LDS --
  {
    const bf16_t* gh = Thi + (size_t)(tb + wave * 16) * CD;
    const bf16_t* gl = Tlo + (size_t)(tb + wave * 16) * CD;
    const unsigned lbh = (unsigned)(size_t)(&sAhi[wave][0]);
    const unsigned lbl = (unsigned)(size_t)(&sAlo[wave][0]);
    // 16*CD bf16 = 16KB per array = 1024 x b128; 32 lanes -> 32 issues each
    for (int i = lane; i < (16 * CD) / 8; i += 32) {
      const unsigned goff = (unsigned)i * 16u;
      asm volatile("global_load_async_to_lds_b128 %0, %1, %2"
                   :: "v"(lbh + goff), "v"(goff), "s"(gh) : "memory");
      asm volatile("global_load_async_to_lds_b128 %0, %1, %2"
                   :: "v"(lbl + goff), "v"(goff), "s"(gl) : "memory");
    }
    asm volatile("s_wait_asynccnt 0" ::: "memory");
  }
  __syncthreads();

  const int m     = lane & 15;
  const int kAoff = (lane >> 4) << 4;
  const int kBoff = (lane >> 4) << 4;

  float bestv[8];
  int   besti[8];
#pragma unroll
  for (int r = 0; r < 8; ++r) { bestv[r] = 3.4e38f; besti[r] = 0; }

  // all 4 waves scan the SAME codes in the same order -> WGP$ reuse of B
  for (int grp = 0; grp < NC / 64; ++grp) {
    const int cbase = grp * 64;

    v8f acc[4];
#pragma unroll
    for (int g = 0; g < 4; ++g)
      acc[g] = (v8f){0.f, 0.f, 0.f, 0.f, 0.f, 0.f, 0.f, 0.f};

#pragma unroll 4
    for (int kc = 0; kc < CD / 32; ++kc) {
      const int k0 = kc * 32;
      v16bf ah, al;
      {
        const bf16_t* p = &sAhi[wave][m * CD + k0 + kAoff];
        *(uint4*)&ah       = *(const uint4*)(p);
        *((uint4*)&ah + 1) = *(const uint4*)(p + 8);
        const bf16_t* q = &sAlo[wave][m * CD + k0 + kAoff];
        *(uint4*)&al       = *(const uint4*)(q);
        *((uint4*)&al + 1) = *(const uint4*)(q + 8);
      }
#pragma unroll
      for (int g = 0; g < 4; ++g) {
        const bf16_t* ph = CBhi + (size_t)(cbase + g * 16 + m) * CD + k0 + kBoff;
        const bf16_t* pl = CBlo + (size_t)(cbase + g * 16 + m) * CD + k0 + kBoff;
        v16bf bh, bl;
        *(uint4*)&bh       = *(const uint4*)(ph);
        *((uint4*)&bh + 1) = *(const uint4*)(ph + 8);
        *(uint4*)&bl       = *(const uint4*)(pl);
        *((uint4*)&bl + 1) = *(const uint4*)(pl + 8);
        acc[g] = wmma_bf16(ah, bh, acc[g]);
        acc[g] = wmma_bf16(ah, bl, acc[g]);
        acc[g] = wmma_bf16(al, bh, acc[g]);
      }
    }

    // pull next group's codebook lines toward the caches
    if (grp + 1 < NC / 64) {
      __builtin_prefetch(CBhi + (size_t)(cbase + 64 + m) * CD, 0, 1);
      __builtin_prefetch(CBlo + (size_t)(cbase + 64 + m) * CD, 0, 1);
    }

    // fused argmin epilogue: d = c2 - 2*(t.c); lane owns column cbase+g*16+m
#pragma unroll
    for (int g = 0; g < 4; ++g) {
      const int   code = cbase + g * 16 + m;
      const float cc   = c2[code];
#pragma unroll
      for (int r = 0; r < 8; ++r) {
        const float d = cc - 2.f * acc[g][r];
        if (d < bestv[r]) { bestv[r] = d; besti[r] = code; }
      }
    }
  }

  // in-wave reduction: row = 8*(lane>=16)+r owns 16 column-lane candidates
  {
    const int half = (lane >> 4) << 3;
#pragma unroll
    for (int r = 0; r < 8; ++r) {
      sVal[wave][(half + r) * 16 + m] = bestv[r];
      sIdx[wave][(half + r) * 16 + m] = besti[r];
    }
  }
  __syncthreads();

  if (lane < 16) {
    float bv = 3.4e38f;
    int   bi = 0x7fffffff;
    for (int s = 0; s < 16; ++s) {
      float v = sVal[wave][lane * 16 + s];
      int   i = sIdx[wave][lane * 16 + s];
      if (v < bv || (v == bv && i < bi)) { bv = v; bi = i; }  // argmin-first ties
    }
    out[tb + wave * 16 + lane] = (float)bi;
  }
}

// ---------------------------------------------------------------------------
extern "C" void kernel_launch(void* const* d_in, const int* in_sizes, int n_in,
                              void* d_out, int out_size, void* d_ws, size_t ws_size,
                              hipStream_t stream) {
  (void)in_sizes; (void)n_in; (void)out_size; (void)ws_size;
  const float* X  = (const float*)d_in[0];   // (8,1024,320) fp32
  // d_in[1]: mask_time_indices, all ones -> N static, unused
  const float* W  = (const float*)d_in[2];   // (512,320) fp32
  const float* CB = (const float*)d_in[3];   // (8192,512) fp32
  float* out = (float*)d_out;                // 8192 argmin indices as fp32

  char*  ws  = (char*)d_ws;
  size_t off = 0;
  auto alloc = [&](size_t bytes) -> void* {
    void* p = ws + off;
    off += (bytes + 255) & ~(size_t)255;
    return p;
  };
  bf16_t* Xhi  = (bf16_t*)alloc((size_t)TOKENS * DIN * 2);
  bf16_t* Xlo  = (bf16_t*)alloc((size_t)TOKENS * DIN * 2);
  bf16_t* Whi  = (bf16_t*)alloc((size_t)CD * DIN * 2);
  bf16_t* Wlo  = (bf16_t*)alloc((size_t)CD * DIN * 2);
  bf16_t* CBhi = (bf16_t*)alloc((size_t)NC * CD * 2);
  bf16_t* CBlo = (bf16_t*)alloc((size_t)NC * CD * 2);
  float*  c2n  = (float*)alloc((size_t)NC * 4);
  bf16_t* Thi  = (bf16_t*)alloc((size_t)TOKENS * CD * 2);
  bf16_t* Tlo  = (bf16_t*)alloc((size_t)TOKENS * CD * 2);

  const int nX = TOKENS * DIN, nW = CD * DIN, nC = NC * CD;
  rpq_split<<<(nX + 255) / 256, 256, 0, stream>>>(X, Xhi, Xlo, nX);
  rpq_split<<<(nW + 255) / 256, 256, 0, stream>>>(W, Whi, Wlo, nW);
  rpq_split<<<(nC + 255) / 256, 256, 0, stream>>>(CB, CBhi, CBlo, nC);
  rpq_c2<<<(NC + 255) / 256, 256, 0, stream>>>(CB, c2n);

  rpq_proj<<<dim3(CD / 16, TOKENS / 16), 32, 0, stream>>>(Xhi, Xlo, Whi, Wlo,
                                                          Thi, Tlo);
  rpq_dist_argmin<<<TOKENS / 64, 128, 0, stream>>>(Thi, Tlo, CBhi, CBlo, c2n,
                                                   out);
}
